// CausalMed_69252052680784
// MI455X (gfx1250) — matrix operand order repaired and checked by reading, not compiled, not run on hardware
//
#include <hip/hip_runtime.h>
#include <hip/hip_bf16.h>

// ---------------------------------------------------------------------------
// CausalMed on MI455X (gfx1250): bf16 WMMA (16x16x32, f32 accum) for all
// K>=256 contractions; set-dim sums collapsed analytically; weights
// pre-converted once to bf16 Bt layout; LDS weight tiles staged with the
// Tensor Data Mover (tensor_load_to_lds + s_wait_tensorcnt) when available.
// ---------------------------------------------------------------------------

typedef __bf16 bf16;
typedef __attribute__((ext_vector_type(16))) __bf16 v16bf;
typedef __attribute__((ext_vector_type(8)))  float  v8f;
typedef __attribute__((ext_vector_type(4)))  unsigned int u32x4;
typedef __attribute__((ext_vector_type(8)))  int i32x8;
typedef __attribute__((ext_vector_type(4)))  int i32x4;

#ifndef __has_builtin
#define __has_builtin(x) 0
#endif
#if __has_builtin(__builtin_amdgcn_tensor_load_to_lds) && \
    __has_builtin(__builtin_amdgcn_s_wait_tensorcnt)
#define HAVE_TDM 1
#else
#define HAVE_TDM 0
#endif

union ABFrag { v16bf v; uint4 q[2]; bf16 h[16]; };

#define WMMA_BF16(a, b, c) \
  __builtin_amdgcn_wmma_f32_16x16x32_bf16(false, (a), false, (b), (short)0, (c), false, false)

// A fragment (16x32 bf16, row-major [M][K], stride in elements).
// Lane L: M = m0 + L%16; elems 0..7 -> K = k0 + 8*(L/16) + i,
// elems 8..15 -> K = k0 + 16 + 8*(L/16) + i  (ISA 7.12.2 16-bit A layout).
__device__ __forceinline__ v16bf load_a_bf(const bf16* base, int m0, int k0,
                                           int stride, int lane) {
  int row = m0 + (lane & 15);
  const bf16* p = base + (size_t)row * stride + k0 + ((lane >> 4) * 8);
  ABFrag f;
  f.q[0] = *(const uint4*)(p);
  f.q[1] = *(const uint4*)(p + 16);
  return f.v;
}

// B fragment from Bt storage (Bt[n][k] = B[k][n], row-major, stride elems).
// Lane L: N = n0 + L%16; elems 0..7 -> K = k0 + 8*(L/16) + e,
// elems 8..15 -> K = k0 + 16 + 8*(L/16) + (e-8).
__device__ __forceinline__ v16bf load_bt_bf(const bf16* base, int n0, int k0,
                                            int stride, int lane) {
  int n = n0 + (lane & 15);
  const bf16* p = base + (size_t)n * stride + k0 + ((lane >> 4) * 8);
  ABFrag f;
  f.q[0] = *(const uint4*)(p);
  f.q[1] = *(const uint4*)(p + 16);
  return f.v;
}

__device__ __forceinline__ v8f vzero8() {
  v8f z = {0.f, 0.f, 0.f, 0.f, 0.f, 0.f, 0.f, 0.f};
  return z;
}

// ---------------------------------------------------------------------------
// TDM staging of a [256 rows x 32 bf16] weight chunk (tensor stride 256)
// into a packed LDS tile wc[256][32].  D# per ISA 08_async_tensor §8.3/8.4:
//   g0: count=1 | lds_addr | global_addr | type=2
//   g1: data_size=2B, tensor_dim0=256, tensor_dim1=256, tile=32x256, stride=256
// ---------------------------------------------------------------------------
#if HAVE_TDM
__device__ __forceinline__ void tdm_stage_wc(bf16* wc, const bf16* gsrc) {
  unsigned lds = (unsigned)(unsigned long long)(void*)wc;
  unsigned long long ga = (unsigned long long)gsrc;
  u32x4 g0 = {1u, lds, (unsigned)ga,
              (unsigned)((ga >> 32) & 0x1FFFFFFu) | (2u << 30)};
  i32x8 g1 = {(int)0x10000u,          // data_size = 1 (2 bytes)
              (int)(256u << 16),      // tensor_dim0 = 256 (low 16 at bit 48)
              (int)(256u << 16),      // tensor_dim1 = 256 (low 16 at bit 80)
              (int)(32u << 16),       // tile_dim0 = 32 (bits 127:112)
              256,                    // tile_dim1 = 256 (bits 143:128)
              256,                    // tensor_dim0_stride = 256 (bits 191:160)
              0, 0};
  i32x4 z4 = {0, 0, 0, 0};
#if (__clang_major__ < 23)
  __builtin_amdgcn_tensor_load_to_lds(g0, g1, z4, z4, 0);
#else
  i32x8 z8 = {0, 0, 0, 0, 0, 0, 0, 0};
  __builtin_amdgcn_tensor_load_to_lds(g0, g1, z4, z4, z8, 0);
#endif
  __builtin_amdgcn_s_wait_tensorcnt(0);
}
#endif

// Stage one K-chunk of a Bt-layout bf16 weight matrix into LDS wc[256][32].
// Wt points at the matrix (rows of 256 elems); chunk kc covers cols kc*32..+32.
__device__ __forceinline__ void stage_wchunk(bf16* wc, const bf16* Wt, int kc,
                                             int tid, int w) {
#if HAVE_TDM
  if (w == 0) tdm_stage_wc(wc, Wt + kc * 32);
#else
  const uint4* src = (const uint4*)Wt;
  uint4* dst = (uint4*)wc;
  for (int i = tid; i < 1024; i += 256) {
    int e = i >> 2, seg = i & 3;              // row e, 16B segment
    dst[i] = src[e * 32 + kc * 4 + seg];
  }
#endif
}

// ---------------------------------------------------------------------------
// Stage 0: init GRU hidden state + batch_neg accumulator
// ---------------------------------------------------------------------------
__global__ __launch_bounds__(256) void init_kernel(float* Hf, bf16* Hbf, float* out) {
  int i = blockIdx.x * 256 + threadIdx.x;
  if (i < 3 * 1024 * 256) { Hf[i] = 0.f; Hbf[i] = (bf16)0.f; }
  if (i == 0) out[1024 * 131] = 0.f;
}

// Prep: dst[e][d] = (bf16)src[d][e] for a 256x256 matrix (Bt layout).
__global__ __launch_bounds__(256) void conv_transpose256(const float* __restrict__ src,
                                                         bf16* __restrict__ dst) {
  int i = blockIdx.x * 256 + threadIdx.x;      // 65536 elements
  int e = i >> 8, d = i & 255;
  dst[(size_t)e * 256 + d] = (bf16)src[(size_t)d * 256 + e];
}

// Prep: flat f32 -> bf16 copy.
__global__ __launch_bounds__(256) void conv_copy(const float* __restrict__ src,
                                                 bf16* __restrict__ dst, int n) {
  int i = blockIdx.x * 256 + threadIdx.x;
  if (i < n) dst[i] = (bf16)src[i];
}

// Prep: Wq [131][1536] -> bf16 [144][1536] with zero padding rows.
__global__ __launch_bounds__(256) void conv_wq(const float* __restrict__ Wq,
                                               bf16* __restrict__ dst) {
  int i = blockIdx.x * 256 + threadIdx.x;      // 144*1536 = 221184
  int row = i / 1536, col = i % 1536;
  float v = (row < 131) ? Wq[(size_t)row * 1536 + col] : 0.f;
  dst[i] = (bf16)v;
}

// ---------------------------------------------------------------------------
// Stage 1: per-(b,v) homo GAT. One workgroup per (b,v); sets padded to 32.
// seq[rel][bv][e] = rho[rel,0] * (colsum softmax(mask(h h^T/16))) . h
// ---------------------------------------------------------------------------
__global__ __launch_bounds__(256) void homo_kernel(
    const int* __restrict__ idx, const float* __restrict__ E,
    const float* __restrict__ adj, const bf16* __restrict__ Wt,
    const float* __restrict__ rho, float* __restrict__ seq,
    int N, int relidx, int prevshift) {
  __shared__ bf16  xs[32 * 256];   // x rows (bf16), reused as h rows after GEMM1
  __shared__ bf16  wc[256 * 32];   // Bt weight chunk: wc[e][kk]
  __shared__ float sm[32 * 32];    // attention scores / probs
  __shared__ float gvec[32];       // column sums of attention probs

  int bv = blockIdx.x, b = bv >> 3, v = bv & 7;
  int tid = threadIdx.x, lane = tid & 31, w = tid >> 5;
  int srcv = prevshift ? (v - 1) : v;

  // gather embeddings -> LDS bf16 pairs (zero pad rows >= N / v==0 history)
  for (int i = tid; i < 32 * 128; i += 256) {
    int row = i >> 7, c2 = (i & 127) * 2;
    float v0 = 0.f, v1 = 0.f;
    if (row < N && srcv >= 0) {
      int tok = idx[(b * 8 + srcv) * N + row];
      const float* ep = E + (size_t)tok * 256 + c2;
      v0 = ep[0]; v1 = ep[1];
    }
    union { bf16 h[2]; unsigned u; } pk;
    pk.h[0] = (bf16)v0; pk.h[1] = (bf16)v1;
    *(unsigned*)&xs[row * 256 + c2] = pk.u;
  }
  __syncthreads();

  // h = x @ W : wave w owns M-tile (w&1) and 4 N-tiles starting at (w>>1)*4
  int mt = w & 1;
  int et0 = (w >> 1) * 4;
  v8f acc[4];
#pragma unroll
  for (int j = 0; j < 4; ++j) acc[j] = vzero8();

  for (int kc = 0; kc < 8; ++kc) {
    stage_wchunk(wc, Wt, kc, tid, w);
    __syncthreads();
    v16bf a = load_a_bf(xs, mt * 16, kc * 32, 256, lane);
#pragma unroll
    for (int j = 0; j < 4; ++j) {
      v16bf bb = load_bt_bf(wc, (et0 + j) * 16, 0, 32, lane);
      acc[j] = WMMA_BF16(a, bb, acc[j]);
    }
    __syncthreads();
  }

  // write h (bf16) back over xs  (C/D layout: lane=N, vgpr r -> M = r + 8*(L/16))
#pragma unroll
  for (int j = 0; j < 4; ++j) {
    int e0 = (et0 + j) * 16 + (lane & 15);
#pragma unroll
    for (int r = 0; r < 8; ++r) {
      int m = mt * 16 + r + 8 * (lane >> 4);
      xs[m * 256 + e0] = (bf16)acc[j][r];
    }
  }
  __syncthreads();

  // s = (h h^T) / sqrt(D): 2x2 tiles, waves 0..3
  if (w < 4) {
    v8f sc = vzero8();
    int smt = w & 1, snt = w >> 1;
    for (int kc = 0; kc < 8; ++kc) {
      v16bf a  = load_a_bf(xs, smt * 16, kc * 32, 256, lane);
      v16bf bb = load_bt_bf(xs, snt * 16, kc * 32, 256, lane);
      sc = WMMA_BF16(a, bb, sc);
    }
    int n0 = snt * 16 + (lane & 15);
#pragma unroll
    for (int r = 0; r < 8; ++r) {
      int m = smt * 16 + r + 8 * (lane >> 4);
      sm[m * 32 + n0] = sc[r] * 0.0625f;  // 1/sqrt(256)
    }
  }
  __syncthreads();

  // masked softmax, one row per thread (rows < N)
  if (tid < 32) {
    int n = tid;
    if (n < N) {
      const float* arow = adj + (size_t)bv * N * N + (size_t)n * N;
      float vals[32];
      float mx = -1e30f;
#pragma unroll
      for (int m2 = 0; m2 < 32; ++m2) {
        bool ok = (m2 < N) && ((m2 == n) || (arow[(m2 < N) ? m2 : 0] > 0.5f));
        float sv = ok ? sm[n * 32 + m2] : -1e9f;
        vals[m2] = sv;
        if (m2 < N && sv > mx) mx = sv;
      }
      float sum = 0.f;
#pragma unroll
      for (int m2 = 0; m2 < 32; ++m2) {
        float e = (m2 < N) ? __expf(vals[m2] - mx) : 0.f;
        vals[m2] = e;
        sum += e;
      }
      float inv = 1.f / sum;
#pragma unroll
      for (int m2 = 0; m2 < 32; ++m2) sm[n * 32 + m2] = vals[m2] * inv;
    } else {
#pragma unroll
      for (int m2 = 0; m2 < 32; ++m2) sm[n * 32 + m2] = 0.f;
    }
  }
  __syncthreads();

  if (tid < 32) {
    float gs = 0.f;
    for (int n = 0; n < N; ++n) gs += sm[n * 32 + tid];
    gvec[tid] = gs;
  }
  __syncthreads();

  // seq[rel][bv][e] = rho0 * sum_m g[m] * h[m][e]
  float rho0 = rho[relidx * 2 + 0];
  float s2 = 0.f;
  for (int m = 0; m < 32; ++m) s2 += gvec[m] * (float)xs[m * 256 + tid];
  seq[((size_t)relidx * 8192 + bv) * 256 + tid] = rho0 * s2;
}

// ---------------------------------------------------------------------------
// Stage 2a: hetero aggregation vectors (set-dim collapsed):
//   t_d  = colsum_n(rownorm(w_dm))   . em_prev
//   t_p  = colsum_n(rownorm(w_pm))   . em_prev
//   t_md = colsum_m(rownorm(w_dm^T)) . ed,   t_mp likewise with w_pm^T, ep
// ---------------------------------------------------------------------------
__global__ __launch_bounds__(256) void hetero_t_kernel(
    const int* __restrict__ idx_diag, const int* __restrict__ idx_proc,
    const int* __restrict__ idx_med, const float* __restrict__ E_diag,
    const float* __restrict__ E_proc, const float* __restrict__ E_med,
    const float* __restrict__ w_dm, const float* __restrict__ w_pm,
    bf16* __restrict__ Td, bf16* __restrict__ Tp,
    bf16* __restrict__ Tmd, bf16* __restrict__ Tmp) {
  __shared__ float wdm[32 * 24], wpm[16 * 24];
  __shared__ float rs_dm[32], rs_pm[16], cs_dm[24], cs_pm[24];
  __shared__ float u_dm[24], u_pm[24], q_d[32], q_p[16];

  int bv = blockIdx.x, b = bv >> 3, v = bv & 7, tid = threadIdx.x;
  float nf = (v > 0) ? 1.f : 0.f;

  const float* gdm = w_dm + (size_t)bv * 32 * 24;
  for (int i = tid; i < 32 * 24; i += 256) wdm[i] = gdm[i] * nf;
  const float* gpm = w_pm + (size_t)bv * 16 * 24;
  for (int i = tid; i < 16 * 24; i += 256) wpm[i] = gpm[i] * nf;
  __syncthreads();

  const float EPS = 1e-8f;
  if (tid < 32) { float s = 0.f; for (int m = 0; m < 24; ++m) s += wdm[tid * 24 + m]; rs_dm[tid] = s + EPS; }
  if (tid >= 32 && tid < 48) { int n = tid - 32; float s = 0.f; for (int m = 0; m < 24; ++m) s += wpm[n * 24 + m]; rs_pm[n] = s + EPS; }
  if (tid >= 64 && tid < 88) { int m = tid - 64; float s = 0.f; for (int n = 0; n < 32; ++n) s += wdm[n * 24 + m]; cs_dm[m] = s + EPS; }
  if (tid >= 96 && tid < 120) { int m = tid - 96; float s = 0.f; for (int n = 0; n < 16; ++n) s += wpm[n * 24 + m]; cs_pm[m] = s + EPS; }
  __syncthreads();

  if (tid < 24) {
    float s = 0.f;  for (int n = 0; n < 32; ++n) s  += wdm[n * 24 + tid] / rs_dm[n];
    float s2 = 0.f; for (int n = 0; n < 16; ++n) s2 += wpm[n * 24 + tid] / rs_pm[n];
    u_dm[tid] = s; u_pm[tid] = s2;
  }
  if (tid >= 32 && tid < 64) { int n = tid - 32; float s = 0.f; for (int m = 0; m < 24; ++m) s += wdm[n * 24 + m] / cs_dm[m]; q_d[n] = s; }
  if (tid >= 64 && tid < 80) { int n = tid - 64; float s = 0.f; for (int m = 0; m < 24; ++m) s += wpm[n * 24 + m] / cs_pm[m]; q_p[n] = s; }
  __syncthreads();

  int e = tid;
  float td = 0.f, tp = 0.f;
  if (v > 0) {
    const int* imrow = idx_med + (b * 8 + (v - 1)) * 24;
    for (int m = 0; m < 24; ++m) {
      float em = E_med[(size_t)imrow[m] * 256 + e];
      td += u_dm[m] * em;
      tp += u_pm[m] * em;
    }
  }
  float tmd = 0.f, tmpv = 0.f;
  const int* idr = idx_diag + (size_t)bv * 32;
  for (int n = 0; n < 32; ++n) tmd += q_d[n] * E_diag[(size_t)idr[n] * 256 + e];
  const int* ipr = idx_proc + (size_t)bv * 16;
  for (int n = 0; n < 16; ++n) tmpv += q_p[n] * E_proc[(size_t)ipr[n] * 256 + e];

  size_t o = (size_t)bv * 256 + e;
  Td[o] = (bf16)td; Tp[o] = (bf16)tp; Tmd[o] = (bf16)tmd; Tmp[o] = (bf16)tmpv;
}

// ---------------------------------------------------------------------------
// Stage 2b: seq[rel] += rho[rel,1] * T @ Wh   ([8192x256]@[256x256] WMMA GEMM)
// WhT is the pre-converted bf16 Bt-layout weight; chunks staged via TDM.
// ---------------------------------------------------------------------------
__global__ __launch_bounds__(256) void gemm_tw_acc(
    const bf16* __restrict__ T, const bf16* __restrict__ WhT,
    const float* __restrict__ rho, int relidx, float* __restrict__ seqr) {
  __shared__ bf16 wc[256 * 32];
  int tid = threadIdx.x, lane = tid & 31, w = tid >> 5;
  int m0 = blockIdx.x * 32;
  int mt = w & 1, et0 = (w >> 1) * 4;
  v8f acc[4];
#pragma unroll
  for (int j = 0; j < 4; ++j) acc[j] = vzero8();

  for (int kc = 0; kc < 8; ++kc) {
    stage_wchunk(wc, WhT, kc, tid, w);
    __syncthreads();
    v16bf a = load_a_bf(T, m0 + mt * 16, kc * 32, 256, lane);
#pragma unroll
    for (int j = 0; j < 4; ++j) {
      v16bf bb = load_bt_bf(wc, (et0 + j) * 16, 0, 32, lane);
      acc[j] = WMMA_BF16(a, bb, acc[j]);
    }
    __syncthreads();
  }

  float r1 = rho[relidx * 2 + 1];
#pragma unroll
  for (int j = 0; j < 4; ++j) {
    int e0 = (et0 + j) * 16 + (lane & 15);
#pragma unroll
    for (int r = 0; r < 8; ++r) {
      int m = m0 + mt * 16 + r + 8 * (lane >> 4);
      seqr[(size_t)m * 256 + e0] += r1 * acc[j][r];
    }
  }
}

// ---------------------------------------------------------------------------
// Stage 3: GRU. y==0 -> G = x_t @ wih^T, y==1 -> Gh = h @ whh^T.
// wih/whh bf16 row-major [768][256] are exactly the Bt[N][K] layout.
// ---------------------------------------------------------------------------
__global__ __launch_bounds__(256) void gru_gemm(
    const bf16* __restrict__ seqbf, const bf16* __restrict__ Hbf,
    const bf16* __restrict__ wihbf, const bf16* __restrict__ whhbf,
    int relidx, int t, float* __restrict__ G, float* __restrict__ Gh) {
  int tid = threadIdx.x, lane = tid & 31, w = tid >> 5;
  int m0 = blockIdx.x * 16;
  bool isH = (blockIdx.y == 1);
  const bf16* Bmat = (isH ? whhbf : wihbf) + (size_t)relidx * 768 * 256;
  const bf16* Ax = seqbf + ((size_t)relidx * 8192 + t) * 256;  // batch stride 2048
  const bf16* Ah = Hbf + (size_t)relidx * 1024 * 256;
  float* out = isH ? Gh : G;

  v8f acc[6];
#pragma unroll
  for (int j = 0; j < 6; ++j) acc[j] = vzero8();

  for (int kc = 0; kc < 8; ++kc) {
    v16bf a = isH ? load_a_bf(Ah, m0, kc * 32, 256, lane)
                  : load_a_bf(Ax, m0, kc * 32, 2048, lane);
#pragma unroll
    for (int j = 0; j < 6; ++j) {
      v16bf bb = load_bt_bf(Bmat, (w * 6 + j) * 16, kc * 32, 256, lane);
      acc[j] = WMMA_BF16(a, bb, acc[j]);
    }
  }
#pragma unroll
  for (int j = 0; j < 6; ++j) {
    int col = (w * 6 + j) * 16 + (lane & 15);
#pragma unroll
    for (int r = 0; r < 8; ++r) {
      int m = m0 + r + 8 * (lane >> 4);
      out[(size_t)m * 768 + col] = acc[j][r];
    }
  }
}

__global__ __launch_bounds__(256) void gru_gate(
    const float* __restrict__ G, const float* __restrict__ Gh,
    const float* __restrict__ bih, const float* __restrict__ bhh,
    int relidx, float* __restrict__ Hf, bf16* __restrict__ Hbf) {
  int b = blockIdx.x, d = threadIdx.x;
  const float* bi = bih + relidx * 768;
  const float* bh = bhh + relidx * 768;
  float ir = G[(size_t)b * 768 + d]       + bi[d];
  float iz = G[(size_t)b * 768 + 256 + d] + bi[256 + d];
  float in = G[(size_t)b * 768 + 512 + d] + bi[512 + d];
  float hr = Gh[(size_t)b * 768 + d]       + bh[d];
  float hz = Gh[(size_t)b * 768 + 256 + d] + bh[256 + d];
  float hn = Gh[(size_t)b * 768 + 512 + d] + bh[512 + d];
  float r = 1.f / (1.f + __expf(-(ir + hr)));
  float z = 1.f / (1.f + __expf(-(iz + hz)));
  float n = tanhf(in + r * hn);
  size_t o = ((size_t)relidx * 1024 + b) * 256 + d;
  float h = Hf[o];
  float hnew = (1.f - z) * n + z * h;
  Hf[o] = hnew;
  Hbf[o] = (bf16)hnew;
}

// ---------------------------------------------------------------------------
// Stage 4: patient_repr = relu([hd,hp,hm,hd,hp,hm]); score = pr @ Wq^T + bq
// (od[:,-1] == hlast for a GRU, so the concat is the hidden states doubled)
// ---------------------------------------------------------------------------
__global__ __launch_bounds__(256) void build_pr(const float* __restrict__ Hf,
                                                bf16* __restrict__ PR) {
  int i = blockIdx.x * 256 + threadIdx.x;   // over 1024*1536
  int b = i / 1536, c = i % 1536;
  int cc = c % 768;
  int r = cc / 256, d = cc % 256;
  float v = Hf[((size_t)r * 1024 + b) * 256 + d];
  PR[i] = (bf16)(v > 0.f ? v : 0.f);
}

__global__ __launch_bounds__(256) void score_gemm(
    const bf16* __restrict__ PR, const bf16* __restrict__ Wqbf,
    const float* __restrict__ bq, float* __restrict__ out) {
  int tid = threadIdx.x, lane = tid & 31, w = tid >> 5;
  int m0 = blockIdx.x * 16;
  for (int nt = w; nt < 9; nt += 8) {        // 9 N-tiles cover 131 (padded 144)
    v8f acc = vzero8();
    int n = nt * 16 + (lane & 15);
    for (int kc = 0; kc < 48; ++kc) {
      v16bf a  = load_a_bf(PR, m0, kc * 32, 1536, lane);
      v16bf bb = load_bt_bf(Wqbf, nt * 16, kc * 32, 1536, lane);
      acc = WMMA_BF16(a, bb, acc);
    }
    if (n < 131) {
      float bqv = bq[n];
#pragma unroll
      for (int r = 0; r < 8; ++r) {
        int m = m0 + r + 8 * (lane >> 4);
        out[(size_t)m * 131 + n] = acc[r] + bqv;
      }
    }
  }
}

// ---------------------------------------------------------------------------
// Stage 5: p = sigmoid(score); batch_neg = 5e-4 * sum_b p_b^T DDI p_b
// ---------------------------------------------------------------------------
__global__ __launch_bounds__(256) void sigmoid_kernel(const float* __restrict__ score,
                                                      float* __restrict__ P) {
  int i = blockIdx.x * 256 + threadIdx.x;    // exactly 1024*131 = 524*256
  P[i] = 1.f / (1.f + __expf(-score[i]));
}

__global__ __launch_bounds__(256) void ddi_kernel(const float* __restrict__ P,
                                                  const float* __restrict__ ddi,
                                                  float* __restrict__ out) {
  __shared__ float pp[131];
  __shared__ float red[256];
  int b = blockIdx.x, tid = threadIdx.x;
  if (tid < 131) pp[tid] = P[(size_t)b * 131 + tid];
  __syncthreads();
  float a = 0.f;
  if (tid < 131) {
    float s = 0.f;
    for (int j = 0; j < 131; ++j) s += ddi[tid * 131 + j] * pp[j];
    a = pp[tid] * s;
  }
  red[tid] = a;
  __syncthreads();
  for (int off = 128; off > 0; off >>= 1) {
    if (tid < off) red[tid] += red[tid + off];
    __syncthreads();
  }
  if (tid == 0) atomicAdd(&out[1024 * 131], 0.0005f * red[0]);
}

// ---------------------------------------------------------------------------
extern "C" void kernel_launch(void* const* d_in, const int* in_sizes, int n_in,
                              void* d_out, int out_size, void* d_ws, size_t ws_size,
                              hipStream_t stream) {
  const int*   idx_diag = (const int*)d_in[0];
  const int*   idx_proc = (const int*)d_in[1];
  const int*   idx_med  = (const int*)d_in[2];
  const float* adj_diag = (const float*)d_in[3];
  const float* adj_proc = (const float*)d_in[4];
  const float* adj_med  = (const float*)d_in[5];
  const float* w_dm     = (const float*)d_in[6];
  const float* w_pm     = (const float*)d_in[7];
  const float* E_diag   = (const float*)d_in[8];
  const float* E_proc   = (const float*)d_in[9];
  const float* E_med    = (const float*)d_in[10];
  const float* W_homo   = (const float*)d_in[11];
  const float* Wh_d     = (const float*)d_in[12];
  const float* Wh_p     = (const float*)d_in[13];
  const float* Wh_md    = (const float*)d_in[14];
  const float* Wh_mp    = (const float*)d_in[15];
  const float* rho      = (const float*)d_in[16];
  const float* gru_wih  = (const float*)d_in[17];
  const float* gru_whh  = (const float*)d_in[18];
  const float* gru_bih  = (const float*)d_in[19];
  const float* gru_bhh  = (const float*)d_in[20];
  const float* Wq       = (const float*)d_in[21];
  const float* bq       = (const float*)d_in[22];
  const float* ddi      = (const float*)d_in[23];
  float* out = (float*)d_out;

  // workspace carve-up
  char* ws = (char*)d_ws;
  float* seq   = (float*)ws; ws += (size_t)3 * 8192 * 256 * 4;   // 25.2 MB
  bf16* seqbf  = (bf16*)ws;  ws += (size_t)3 * 8192 * 256 * 2;   // 12.6 MB
  bf16* Td  = (bf16*)ws;  ws += (size_t)8192 * 256 * 2;
  bf16* Tp  = (bf16*)ws;  ws += (size_t)8192 * 256 * 2;
  bf16* Tmd = (bf16*)ws;  ws += (size_t)8192 * 256 * 2;
  bf16* Tmp = (bf16*)ws;  ws += (size_t)8192 * 256 * 2;
  float* G  = (float*)ws; ws += (size_t)1024 * 768 * 4;
  float* Gh = (float*)ws; ws += (size_t)1024 * 768 * 4;
  float* Hf = (float*)ws; ws += (size_t)3 * 1024 * 256 * 4;
  bf16* Hbf = (bf16*)ws;  ws += (size_t)3 * 1024 * 256 * 2;
  bf16* PR  = (bf16*)ws;  ws += (size_t)1024 * 1536 * 2;
  float* P  = (float*)ws; ws += (size_t)1024 * 131 * 4;
  bf16* WtH = (bf16*)ws;  ws += (size_t)3 * 65536 * 2;  // W_homo^T bf16
  bf16* WhT = (bf16*)ws;  ws += (size_t)4 * 65536 * 2;  // Wh_{d,p,md,mp}^T bf16
  bf16* wihbf = (bf16*)ws; ws += (size_t)3 * 768 * 256 * 2;
  bf16* whhbf = (bf16*)ws; ws += (size_t)3 * 768 * 256 * 2;
  bf16* Wqbf  = (bf16*)ws; ws += (size_t)144 * 1536 * 2;

  init_kernel<<<3072, 256, 0, stream>>>(Hf, Hbf, out);

  // one-time weight conversion to bf16 Bt layouts
  for (int r = 0; r < 3; ++r)
    conv_transpose256<<<256, 256, 0, stream>>>(W_homo + (size_t)r * 65536,
                                               WtH + (size_t)r * 65536);
  conv_transpose256<<<256, 256, 0, stream>>>(Wh_d,  WhT + 0 * 65536);
  conv_transpose256<<<256, 256, 0, stream>>>(Wh_p,  WhT + 1 * 65536);
  conv_transpose256<<<256, 256, 0, stream>>>(Wh_md, WhT + 2 * 65536);
  conv_transpose256<<<256, 256, 0, stream>>>(Wh_mp, WhT + 3 * 65536);
  conv_copy<<<2304, 256, 0, stream>>>(gru_wih, wihbf, 3 * 768 * 256);
  conv_copy<<<2304, 256, 0, stream>>>(gru_whh, whhbf, 3 * 768 * 256);
  conv_wq<<<864, 256, 0, stream>>>(Wq, Wqbf);

  homo_kernel<<<8192, 256, 0, stream>>>(idx_diag, E_diag, adj_diag,
                                        WtH + 0 * 65536, rho, seq, 32, 0, 0);
  homo_kernel<<<8192, 256, 0, stream>>>(idx_proc, E_proc, adj_proc,
                                        WtH + 1 * 65536, rho, seq, 16, 1, 0);
  homo_kernel<<<8192, 256, 0, stream>>>(idx_med, E_med, adj_med,
                                        WtH + 2 * 65536, rho, seq, 24, 2, 1);

  hetero_t_kernel<<<8192, 256, 0, stream>>>(idx_diag, idx_proc, idx_med,
                                            E_diag, E_proc, E_med,
                                            w_dm, w_pm, Td, Tp, Tmd, Tmp);

  gemm_tw_acc<<<256, 256, 0, stream>>>(Td,  WhT + 0 * 65536, rho, 0, seq + (size_t)0 * 8192 * 256);
  gemm_tw_acc<<<256, 256, 0, stream>>>(Tp,  WhT + 1 * 65536, rho, 1, seq + (size_t)1 * 8192 * 256);
  gemm_tw_acc<<<256, 256, 0, stream>>>(Tmd, WhT + 2 * 65536, rho, 2, seq + (size_t)2 * 8192 * 256);
  gemm_tw_acc<<<256, 256, 0, stream>>>(Tmp, WhT + 3 * 65536, rho, 2, seq + (size_t)2 * 8192 * 256);

  // seq (f32, fully accumulated) -> bf16 for the GRU A-operand
  conv_copy<<<24576, 256, 0, stream>>>(seq, seqbf, 3 * 8192 * 256);

  for (int t = 0; t < 8; ++t) {
    for (int r = 0; r < 3; ++r) {
      gru_gemm<<<dim3(64, 2), 256, 0, stream>>>(seqbf, Hbf, wihbf, whhbf, r, t, G, Gh);
      gru_gate<<<1024, 256, 0, stream>>>(G, Gh, gru_bih, gru_bhh, r, Hf, Hbf);
    }
  }

  build_pr<<<6144, 256, 0, stream>>>(Hf, PR);
  score_gemm<<<64, 256, 0, stream>>>(PR, Wqbf, bq, out);
  sigmoid_kernel<<<524, 256, 0, stream>>>(out, P);
  ddi_kernel<<<1024, 256, 0, stream>>>(P, ddi, out);
}
